// GNN_7679401525631
// MI455X (gfx1250) — compile-verified, matrix operand order checked by reading, not run on hardware
//
#include <hip/hip_runtime.h>
#include <hip/hip_bf16.h>

#define N_NODES 10000
#define N_EDGES 320000
#define N_GRAPHS 8
#define HID 256
#define HID2 512
#define N_LAYERS 6
#define EPB 32  // edges per block in edge kernel (two 16-row WMMA M-tiles)

typedef __attribute__((ext_vector_type(16))) _Float16 v16h;
typedef __attribute__((ext_vector_type(8)))  _Float16 v8h;
typedef __attribute__((ext_vector_type(8)))  float    v8f;

// Build a 16-half WMMA fragment from two contiguous 8-half (16B) chunks.
// Per ISA 7.12.2: lane L holds K = kbase+0..7 in halves 0..7 and K = kbase+16..23
// in halves 8..15, where kbase includes the (lane/16)*8 split.
__device__ __forceinline__ v16h load_frag(const _Float16* p0, const _Float16* p1) {
    v8h lo = *(const v8h*)p0;
    v8h hi = *(const v8h*)p1;
    v16h r;
#pragma unroll
    for (int i = 0; i < 8; ++i) { r[i] = lo[i]; r[i + 8] = hi[i]; }
    return r;
}

__device__ __forceinline__ void atomicMaxFloat(float* addr, float val) {
    unsigned int* ua = (unsigned int*)addr;
    unsigned int old = *ua;
    while (__uint_as_float(old) < val) {
        unsigned int assumed = old;
        old = atomicCAS(ua, assumed, __float_as_uint(val));
        if (old == assumed) break;
    }
}

// ---------- weight preprocessing: W[K][N] f32 -> Wt[N][K] f16 ----------
__global__ void transpose_cast_kernel(const float* __restrict__ W, _Float16* __restrict__ Wt,
                                      int K, int Nout) {
    int idx = blockIdx.x * blockDim.x + threadIdx.x;
    if (idx >= K * Nout) return;
    int n = idx / K;
    int k = idx - n * K;
    Wt[n * K + k] = (_Float16)W[k * Nout + n];
}

// ---------- degree / inverse degree ----------
__global__ void deg_kernel(const int* __restrict__ dst, float* __restrict__ deg) {
    int e = blockIdx.x * blockDim.x + threadIdx.x;
    if (e < N_EDGES) atomicAdd(&deg[dst[e]], 1.0f);
}

__global__ void inv_deg_kernel(const float* __restrict__ deg, float* __restrict__ invd) {
    int n = blockIdx.x * blockDim.x + threadIdx.x;
    if (n < N_NODES) {
        float d = deg[n];
        invd[n] = (d > 0.0f) ? (1.0f / d) : 0.0f;
    }
}

// ---------- encoder layer 1: [N,3] @ [3,256] + b, ReLU ----------
__global__ void enc1_kernel(const float* __restrict__ x, const float* __restrict__ W1,
                            const float* __restrict__ b1, float* __restrict__ out) {
    int idx = blockIdx.x * blockDim.x + threadIdx.x;
    if (idx >= N_NODES * HID) return;
    int n = idx >> 8;
    int c = idx & 255;
    float v = x[n * 3 + 0] * W1[0 * HID + c]
            + x[n * 3 + 1] * W1[1 * HID + c]
            + x[n * 3 + 2] * W1[2 * HID + c] + b1[c];
    out[idx] = v > 0.0f ? v : 0.0f;
}

// ---------- generic node GEMM: out[N,256] = act(in[N,256] @ W + b), W as Wt[256][256] f16 ----------
__global__ void node_gemm_kernel(const float* __restrict__ in, const _Float16* __restrict__ Wt,
                                 const float* __restrict__ bias, float* __restrict__ out,
                                 int relu) {
    __shared__ _Float16 a_lds[16 * 256];
    int tid = threadIdx.x;
    int n0 = blockIdx.x * 16;
    for (int idx = tid; idx < 16 * 256; idx += 256) {
        int rl = idx >> 8;
        int c  = idx & 255;
        int n  = n0 + rl;
        a_lds[idx] = (n < N_NODES) ? (_Float16)in[n * HID + c] : (_Float16)0.0f;
    }
    __syncthreads();

    int wave = tid >> 5, lane = tid & 31;
    int row16  = lane & 15;
    int khalf  = (lane >> 4) * 8;
    int rowsel = (lane >> 4) * 8;
#pragma unroll
    for (int nt = 0; nt < 2; ++nt) {
        int col = (wave * 2 + nt) * 16 + row16;
        v8f acc = {};
#pragma unroll 4
        for (int kt = 0; kt < 8; ++kt) {
            int kb = kt * 32 + khalf;
            v16h a = load_frag(&a_lds[row16 * 256 + kb], &a_lds[row16 * 256 + kb + 16]);
            v16h b = load_frag(&Wt[col * 256 + kb], &Wt[col * 256 + kb + 16]);
            acc = __builtin_amdgcn_wmma_f32_16x16x32_f16(false, a, false, b,
                                                         (short)0, acc, false, false);
        }
        float bv = bias[col];
#pragma unroll
        for (int r = 0; r < 8; ++r) {
            int row = n0 + rowsel + r;
            if (row < N_NODES) {
                float v = acc[r] + bv;
                if (relu) v = v > 0.0f ? v : 0.0f;
                out[row * HID + col] = v;
            }
        }
    }
}

// ---------- fused EdgeConv MLP ----------
// m = [xi, xj-xi] -> ReLU(m@W1+b1)@W2+b2, then hnext[dst] += out * inv_deg[dst] (atomic).
// 32 edges per block = two 16-row M-tiles sharing each B (weight) fragment:
// halves per-layer weight traffic from L2 vs a 16-edge tile.
__global__ void edge_mlp_kernel(const float* __restrict__ h,
                                const int* __restrict__ src, const int* __restrict__ dst,
                                const _Float16* __restrict__ W1t,  // [256][512] f16
                                const float* __restrict__ b1,
                                const _Float16* __restrict__ W2t,  // [256][256] f16
                                const float* __restrict__ b2,
                                const float* __restrict__ invd,
                                float* __restrict__ hnext) {
    __shared__ _Float16 m_lds[EPB * 512];    // 32 KB
    __shared__ _Float16 h1_lds[EPB * 256];   // 16 KB
    __shared__ int   dst_l[EPB];
    __shared__ float scl_l[EPB];

    int tid = threadIdx.x;
    int e0 = blockIdx.x * EPB;

    if (tid < EPB) {
        int e = e0 + tid;
        int d = (e < N_EDGES) ? dst[e] : -1;
        dst_l[tid] = d;
        scl_l[tid] = (d >= 0) ? invd[d] : 0.0f;
    }
    // stage [32 edges][512] = concat(xi, xj - xi) as f16
    for (int idx = tid; idx < EPB * 512; idx += 256) {
        int el = idx >> 9;
        int c  = idx & 511;
        int e  = e0 + el;
        _Float16 v = (_Float16)0.0f;
        if (e < N_EDGES) {
            int d = dst[e];
            if (c < 256) {
                v = (_Float16)h[d * HID + c];
            } else {
                int s = src[e];
                int cc = c - 256;
                v = (_Float16)(h[s * HID + cc] - h[d * HID + cc]);
            }
        }
        m_lds[idx] = v;
    }
    __syncthreads();

    int wave = tid >> 5, lane = tid & 31;
    int row16  = lane & 15;
    int khalf  = (lane >> 4) * 8;
    int rowsel = (lane >> 4) * 8;

    // layer 1: [32,512] @ [512,256], ReLU -> f16 LDS. B fragment reused by both M-tiles.
#pragma unroll
    for (int nt = 0; nt < 2; ++nt) {
        int col = (wave * 2 + nt) * 16 + row16;
        v8f acc0 = {}, acc1 = {};
#pragma unroll 4
        for (int kt = 0; kt < 16; ++kt) {
            int kb = kt * 32 + khalf;
            v16h b  = load_frag(&W1t[col * 512 + kb], &W1t[col * 512 + kb + 16]);
            v16h a0 = load_frag(&m_lds[row16 * 512 + kb],
                                &m_lds[row16 * 512 + kb + 16]);
            v16h a1 = load_frag(&m_lds[(16 + row16) * 512 + kb],
                                &m_lds[(16 + row16) * 512 + kb + 16]);
            acc0 = __builtin_amdgcn_wmma_f32_16x16x32_f16(false, a0, false, b,
                                                          (short)0, acc0, false, false);
            acc1 = __builtin_amdgcn_wmma_f32_16x16x32_f16(false, a1, false, b,
                                                          (short)0, acc1, false, false);
        }
        float bv = b1[col];
#pragma unroll
        for (int r = 0; r < 8; ++r) {
            float v0 = acc0[r] + bv;
            float v1 = acc1[r] + bv;
            v0 = v0 > 0.0f ? v0 : 0.0f;
            v1 = v1 > 0.0f ? v1 : 0.0f;
            h1_lds[(rowsel + r) * 256 + col]      = (_Float16)v0;
            h1_lds[(16 + rowsel + r) * 256 + col] = (_Float16)v1;
        }
    }
    __syncthreads();

    // layer 2: [32,256] @ [256,256] (+b2), scatter hnext[dst] += val * inv_deg[dst]
#pragma unroll
    for (int nt = 0; nt < 2; ++nt) {
        int col = (wave * 2 + nt) * 16 + row16;
        v8f acc0 = {}, acc1 = {};
#pragma unroll 4
        for (int kt = 0; kt < 8; ++kt) {
            int kb = kt * 32 + khalf;
            v16h b  = load_frag(&W2t[col * 256 + kb], &W2t[col * 256 + kb + 16]);
            v16h a0 = load_frag(&h1_lds[row16 * 256 + kb],
                                &h1_lds[row16 * 256 + kb + 16]);
            v16h a1 = load_frag(&h1_lds[(16 + row16) * 256 + kb],
                                &h1_lds[(16 + row16) * 256 + kb + 16]);
            acc0 = __builtin_amdgcn_wmma_f32_16x16x32_f16(false, a0, false, b,
                                                          (short)0, acc0, false, false);
            acc1 = __builtin_amdgcn_wmma_f32_16x16x32_f16(false, a1, false, b,
                                                          (short)0, acc1, false, false);
        }
        float bv = b2[col];
#pragma unroll
        for (int r = 0; r < 8; ++r) {
            int rl0 = rowsel + r;
            int rl1 = 16 + rowsel + r;
            int d0 = dst_l[rl0];
            int d1 = dst_l[rl1];
            if (d0 >= 0) atomicAdd(&hnext[d0 * HID + col], (acc0[r] + bv) * scl_l[rl0]);
            if (d1 >= 0) atomicAdd(&hnext[d1 * HID + col], (acc1[r] + bv) * scl_l[rl1]);
        }
    }
}

// ---------- head layer 2 (256 -> 1) + per-graph segment max ----------
__global__ void init_out_kernel(float* __restrict__ out) {
    int g = threadIdx.x;
    if (g < N_GRAPHS) out[g] = -3.402823466e+38f;
}

// one wave32 per node: coalesced lane-strided dot + shuffle reduction
__global__ void head2_kernel(const float* __restrict__ hid, const float* __restrict__ W2,
                             const float* __restrict__ b2, const int* __restrict__ batch,
                             float* __restrict__ out) {
    int gw = (blockIdx.x * blockDim.x + threadIdx.x) >> 5;
    int lane = threadIdx.x & 31;
    if (gw >= N_NODES) return;
    const float* row = hid + (size_t)gw * HID;
    float s = 0.0f;
#pragma unroll
    for (int c = lane; c < HID; c += 32) s += row[c] * W2[c];
#pragma unroll
    for (int off = 16; off > 0; off >>= 1) s += __shfl_down(s, off, 32);
    if (lane == 0) atomicMaxFloat(&out[batch[gw]], s + b2[0]);
}

extern "C" void kernel_launch(void* const* d_in, const int* in_sizes, int n_in,
                              void* d_out, int out_size, void* d_ws, size_t ws_size,
                              hipStream_t stream) {
    const float* x      = (const float*)d_in[0];
    const int*   ei     = (const int*)d_in[1];
    const int*   batch  = (const int*)d_in[2];
    const float* encW1  = (const float*)d_in[3];
    const float* encB1  = (const float*)d_in[4];
    const float* encW2  = (const float*)d_in[5];
    const float* encB2  = (const float*)d_in[6];
    const float* convW1 = (const float*)d_in[7];
    const float* convB1 = (const float*)d_in[8];
    const float* convW2 = (const float*)d_in[9];
    const float* convB2 = (const float*)d_in[10];
    const float* headW1 = (const float*)d_in[11];
    const float* headB1 = (const float*)d_in[12];
    const float* headW2 = (const float*)d_in[13];
    const float* headB2 = (const float*)d_in[14];
    float* out = (float*)d_out;

    const int* src = ei;            // edge_index[0]
    const int* dst = ei + N_EDGES;  // edge_index[1]

    // ---- workspace layout (all 256B aligned) ----
    char* ws = (char*)d_ws;
    size_t off = 0;
    auto alloc = [&](size_t bytes) -> void* {
        void* p = ws + off;
        off = (off + bytes + 255) & ~(size_t)255;
        return p;
    };
    float* hA   = (float*)alloc(sizeof(float) * N_NODES * HID);
    float* hB   = (float*)alloc(sizeof(float) * N_NODES * HID);
    float* deg  = (float*)alloc(sizeof(float) * N_NODES);
    float* invd = (float*)alloc(sizeof(float) * N_NODES);
    _Float16* encW2t  = (_Float16*)alloc(sizeof(_Float16) * HID * HID);
    _Float16* headW1t = (_Float16*)alloc(sizeof(_Float16) * HID * HID);
    _Float16* convW1t = (_Float16*)alloc(sizeof(_Float16) * N_LAYERS * HID * HID2);
    _Float16* convW2t = (_Float16*)alloc(sizeof(_Float16) * N_LAYERS * HID * HID);

    const size_t hbytes = sizeof(float) * N_NODES * HID;

    // ---- weight preprocessing: cast+transpose to f16 column-major ----
    transpose_cast_kernel<<<(HID * HID + 255) / 256, 256, 0, stream>>>(encW2, encW2t, HID, HID);
    transpose_cast_kernel<<<(HID * HID + 255) / 256, 256, 0, stream>>>(headW1, headW1t, HID, HID);
    for (int l = 0; l < N_LAYERS; ++l) {
        transpose_cast_kernel<<<(HID2 * HID + 255) / 256, 256, 0, stream>>>(
            convW1 + (size_t)l * HID2 * HID, convW1t + (size_t)l * HID * HID2, HID2, HID);
        transpose_cast_kernel<<<(HID * HID + 255) / 256, 256, 0, stream>>>(
            convW2 + (size_t)l * HID * HID, convW2t + (size_t)l * HID * HID, HID, HID);
    }

    // ---- degrees ----
    hipMemsetAsync(deg, 0, sizeof(float) * N_NODES, stream);
    deg_kernel<<<(N_EDGES + 255) / 256, 256, 0, stream>>>(dst, deg);
    inv_deg_kernel<<<(N_NODES + 255) / 256, 256, 0, stream>>>(deg, invd);

    // ---- encoder: layer1 -> hB (scratch), layer2 -> hA ----
    enc1_kernel<<<(N_NODES * HID + 255) / 256, 256, 0, stream>>>(x, encW1, encB1, hB);
    node_gemm_kernel<<<(N_NODES + 15) / 16, 256, 0, stream>>>(hB, encW2t, encB2, hA, 0);

    // ---- 6 EdgeConv layers, ping-pong hA <-> hB; inv_deg scaling fused into atomics ----
    for (int l = 0; l < N_LAYERS; ++l) {
        const float* hin = (l & 1) ? hB : hA;
        float*       hout = (l & 1) ? hA : hB;
        hipMemcpyAsync(hout, hin, hbytes, hipMemcpyDeviceToDevice, stream);
        edge_mlp_kernel<<<N_EDGES / EPB, 256, 0, stream>>>(
            hin, src, dst,
            convW1t + (size_t)l * HID * HID2, convB1 + (size_t)l * HID,
            convW2t + (size_t)l * HID * HID,  convB2 + (size_t)l * HID,
            invd, hout);
    }
    // after 6 layers (even count) the final state is in hA
    const float* hfin = hA;

    // ---- head: layer1 -> hB (scratch), then 256->1 + segment max ----
    node_gemm_kernel<<<(N_NODES + 15) / 16, 256, 0, stream>>>(hfin, headW1t, headB1, hB, 1);
    init_out_kernel<<<1, N_GRAPHS, 0, stream>>>(out);
    head2_kernel<<<(N_NODES * 32 + 255) / 256, 256, 0, stream>>>(hB, headW2, headB2, batch, out);
}
